// GeneGAT_7026566496593
// MI455X (gfx1250) — compile-verified, matrix operand order checked by reading, not run on hardware
//
#include <hip/hip_runtime.h>
#include <hip/hip_bf16.h>
#include <math.h>

// ---------------------------------------------------------------------------
// GeneGAT on MI455X (gfx1250, wave32).
//  * GEMMs: v_wmma_f32_16x16x32_bf16; A staged in LDS (ds_load_b128 frags),
//    B from pre-transposed bf16 weights (2x global_load_b128 per frag).
//  * Edge phase: CSR (count/scan/fill) then per-(node,head) wave gather --
//    no float atomics; ~350MB of L2-resident traffic instead of ~87M atomics.
// ---------------------------------------------------------------------------

#define NN    20000
#define DD    256
#define HID   256
#define HH    4
#define CC    64
#define NEG   0.2f
#define LNEPS 1e-5f

typedef __attribute__((ext_vector_type(16))) __bf16 v16bf;
typedef __attribute__((ext_vector_type(8)))  __bf16 v8bf;
typedef __attribute__((ext_vector_type(8)))  float  v8f;

__device__ __forceinline__ __bf16 f2bf(float f) { return (__bf16)f; }

__device__ __forceinline__ void edge_sd(const int* __restrict__ ei, int e, int E,
                                        int& s, int& d) {
    if (e < E) { s = ei[e]; d = ei[E + e]; }
    else       { s = e - E; d = e - E; }      // self loops appended
}

// ---------------------------------------------------------------------------
// prep: WT[g][k] = (bf16) W_mat[k][col]  (g = mat*256 + col), zero degree
// ---------------------------------------------------------------------------
__global__ void prep_kernel(const float* __restrict__ Wl1,
                            const float* __restrict__ Wr1,
                            const float* __restrict__ Ws,
                            __bf16* __restrict__ WT, int* __restrict__ deg) {
    int i = blockIdx.x * blockDim.x + threadIdx.x;          // 0 .. 768*256-1
    if (i < 3 * HID * DD) {
        int g = i >> 8, k = i & 255;
        int mat = g >> 8, col = g & 255;
        const float* W = (mat == 0) ? Wl1 : ((mat == 1) ? Wr1 : Ws);
        WT[i] = f2bf(W[k * HID + col]);
    }
    if (i < NN) deg[i] = 0;
}

// ---------------------------------------------------------------------------
// CSR build: degree count, one-block exclusive scan, fill
// ---------------------------------------------------------------------------
__global__ void deg_kernel(const int* __restrict__ ei, int E, int ET,
                           int* __restrict__ deg) {
    int e = blockIdx.x * blockDim.x + threadIdx.x;
    if (e >= ET) return;
    int s, d; edge_sd(ei, e, E, s, d);
    atomicAdd(&deg[d], 1);
}

__global__ __launch_bounds__(1024) void scan_kernel(const int* __restrict__ deg,
                                                    int* __restrict__ rowptr,
                                                    int* __restrict__ woff) {
    __shared__ int part[1024];
    const int t = threadIdx.x;
    const int per = 20;                       // 1000 threads x 20 = 20000
    int local[20];
    int sum = 0;
    if (t < 1000) {
        #pragma unroll
        for (int i = 0; i < per; ++i) { local[i] = deg[t * per + i]; sum += local[i]; }
    }
    part[t] = sum;
    __syncthreads();
    if (t == 0) {
        int run = 0;
        for (int i = 0; i < 1024; ++i) { int v = part[i]; part[i] = run; run += v; }
        rowptr[NN] = run;
    }
    __syncthreads();
    if (t < 1000) {
        int run = part[t];
        #pragma unroll
        for (int i = 0; i < per; ++i) {
            rowptr[t * per + i] = run;
            woff[t * per + i]   = run;
            run += local[i];
        }
    }
}

__global__ void fill_kernel(const int* __restrict__ ei, int E, int ET,
                            int* __restrict__ woff, int* __restrict__ esrc) {
    int e = blockIdx.x * blockDim.x + threadIdx.x;
    if (e >= ET) return;
    int s, d; edge_sd(ei, e, E, s, d);
    int pos = atomicAdd(&woff[d], 1);
    esrc[pos] = s;
}

// ---------------------------------------------------------------------------
// gemm3: xl1 = x@Wl1, xr1 = x@Wr1, xs = x@Ws + bs    (bf16 WMMA, f32 accum)
// block = 16 rows x 128 cols of the 768-wide [Wl1|Wr1|Ws] output; 8 waves,
// one wave = one 16x16 tile, K=256 in 8 WMMAs.
// ---------------------------------------------------------------------------
__global__ __launch_bounds__(256) void gemm3_kernel(
    const float* __restrict__ x, const __bf16* __restrict__ WT,
    const float* __restrict__ bs,
    float* __restrict__ xl1, float* __restrict__ xr1, float* __restrict__ xs) {

    __shared__ __bf16 As[16 * DD];          // 8 KB: 16 rows of x in bf16

    const int m0 = blockIdx.x * 16;
    const int t  = threadIdx.x;

    const int wave = t >> 5;
    const int lane = t & 31;
    const int gcol = blockIdx.y * 128 + wave * 16;   // 0..767
    const int mat  = gcol >> 8;                      // 0:Wl1 1:Wr1 2:Ws
    const int col  = gcol & 255;
    const int n    = lane & 15;                      // B column within tile
    const int half = lane >> 4;
    const int kbase = half * 8;                      // 16-bit frag K base

    const __bf16* wt = WT + (size_t)(gcol + n) * DD; // this lane's B column
    __builtin_prefetch(wt, 0, 3);                    // global_prefetch_b8
    __builtin_prefetch(wt + 128, 0, 3);

    // stage A tile: thread t loads 4x float4, packs to bf16, 64-bit LDS stores
    {
        const int row = t >> 4;
        const int cb  = (t & 15) * 16;
        const float4* xp = (const float4*)(x + (size_t)(m0 + row) * DD + cb);
        #pragma unroll
        for (int i = 0; i < 4; ++i) {
            float4 v = xp[i];
            union { __bf16 h[4]; unsigned long long u; } pk;
            pk.h[0] = f2bf(v.x); pk.h[1] = f2bf(v.y);
            pk.h[2] = f2bf(v.z); pk.h[3] = f2bf(v.w);
            *(unsigned long long*)&As[row * DD + cb + 4 * i] = pk.u;
        }
    }
    __syncthreads();

    v8f c = {};
    for (int k0 = 0; k0 < DD; k0 += 32) {
        // A fragment: elements 0..7 -> K=kbase..+7, 8..15 -> K=kbase+16..+23
        const __bf16* arow = As + (lane & 15) * DD + k0 + kbase;
        v8bf alo = *(const v8bf*)(arow);
        v8bf ahi = *(const v8bf*)(arow + 16);
        v16bf a = __builtin_shufflevector(alo, ahi,
                    0,1,2,3,4,5,6,7,8,9,10,11,12,13,14,15);
        // B fragment: contiguous in transposed bf16 weights
        v8bf blo = *(const v8bf*)(wt + k0 + kbase);
        v8bf bhi = *(const v8bf*)(wt + k0 + kbase + 16);
        v16bf b = __builtin_shufflevector(blo, bhi,
                    0,1,2,3,4,5,6,7,8,9,10,11,12,13,14,15);

        c = __builtin_amdgcn_wmma_f32_16x16x32_bf16(
                false, a, false, b, (short)0, c, false, false);
    }

    const float bias = (mat == 2) ? bs[col + n] : 0.f;
    float* out = (mat == 0) ? xl1 : ((mat == 1) ? xr1 : xs);
    #pragma unroll
    for (int r = 0; r < 8; ++r) {
        const int row = m0 + r + half * 8;   // C/D layout: vgpr r, half -> M+8
        out[(size_t)row * HID + col + n] = c[r] + bias;
    }
}

// ---------------------------------------------------------------------------
// conv1 aggregation: one wave per (node, head). Segment softmax + weighted
// gather, wave32 shuffle reductions, no atomics.
// ---------------------------------------------------------------------------
__global__ __launch_bounds__(256) void agg1_kernel(
    const int* __restrict__ rowptr, const int* __restrict__ esrc,
    const float* __restrict__ xl1, const float* __restrict__ xr1,
    const float* __restrict__ att1,
    float* __restrict__ lscratch, float* __restrict__ acc) {

    const int wid  = blockIdx.x * 8 + (threadIdx.x >> 5);
    const int lane = threadIdx.x & 31;
    if (wid >= NN * HH) return;
    const int nid = wid >> 2, h = wid & 3;
    const int start = rowptr[nid], end = rowptr[nid + 1];

    const float4* xr = (const float4*)(xr1 + (size_t)nid * HID + h * CC);
    const float4* at = (const float4*)(att1 + h * CC);

    // pass 1: logits + segment max
    float mx = -3.0e38f;
    for (int i = start + lane; i < end; i += 32) {
        const int s = esrc[i];
        const float4* xl = (const float4*)(xl1 + (size_t)s * HID + h * CC);
        float lg = 0.f;
        #pragma unroll 4
        for (int cIt = 0; cIt < CC / 4; ++cIt) {
            float4 a = xl[cIt], b = xr[cIt], w = at[cIt];
            float h0 = a.x + b.x; h0 = h0 > 0.f ? h0 : NEG * h0;
            float h1 = a.y + b.y; h1 = h1 > 0.f ? h1 : NEG * h1;
            float h2 = a.z + b.z; h2 = h2 > 0.f ? h2 : NEG * h2;
            float h3 = a.w + b.w; h3 = h3 > 0.f ? h3 : NEG * h3;
            lg += h0 * w.x + h1 * w.y + h2 * w.z + h3 * w.w;
        }
        lscratch[(size_t)i * HH + h] = lg;
        mx = fmaxf(mx, lg);
    }
    #pragma unroll
    for (int off = 16; off > 0; off >>= 1) mx = fmaxf(mx, __shfl_xor(mx, off, 32));

    // pass 2: exp + segment sum (overwrite scratch with p)
    float z = 0.f;
    for (int i = start + lane; i < end; i += 32) {
        float p = __expf(lscratch[(size_t)i * HH + h] - mx);
        lscratch[(size_t)i * HH + h] = p;
        z += p;
    }
    #pragma unroll
    for (int off = 16; off > 0; off >>= 1) z += __shfl_xor(z, off, 32);
    const float inv = 1.f / z;

    // pass 3: weighted gather; lane owns channels (lane, lane+32)
    float a0 = 0.f, a1 = 0.f;
    for (int i = start; i < end; ++i) {
        const float alpha = lscratch[(size_t)i * HH + h] * inv;  // wave-uniform
        const int s = esrc[i];
        const float* xl = xl1 + (size_t)s * HID + h * CC;
        a0 += alpha * xl[lane];
        a1 += alpha * xl[lane + 32];
    }
    acc[(size_t)nid * HID + h * CC + lane]      = a0;
    acc[(size_t)nid * HID + h * CC + lane + 32] = a1;
}

// ---------------------------------------------------------------------------
// per-node: +b1 +skip, LayerNorm, ELU, then xe@Wl2 / xe@Wr2 (256 -> 2 each)
// ---------------------------------------------------------------------------
__device__ __forceinline__ float blockReduceSum256(float v) {
    __shared__ float smem[8];
    __syncthreads();                                  // protect smem reuse
    #pragma unroll
    for (int off = 16; off > 0; off >>= 1) v += __shfl_down(v, off, 32);
    if ((threadIdx.x & 31) == 0) smem[threadIdx.x >> 5] = v;
    __syncthreads();
    float r = 0.f;
    #pragma unroll
    for (int i = 0; i < 8; ++i) r += smem[i];
    return r;
}

__global__ __launch_bounds__(256) void nodepost_kernel(
    const float* __restrict__ acc, const float* __restrict__ xs,
    const float* __restrict__ b1,
    const float* __restrict__ gamma, const float* __restrict__ beta,
    const float* __restrict__ Wl2, const float* __restrict__ Wr2,
    float* __restrict__ xl2, float* __restrict__ xr2) {
    const int nid = blockIdx.x;
    const int t   = threadIdx.x;
    float v = acc[(size_t)nid * HID + t] + b1[t] + xs[(size_t)nid * HID + t];
    float mu  = blockReduceSum256(v) * (1.f / HID);
    float dv  = v - mu;
    float var = blockReduceSum256(dv * dv) * (1.f / HID);
    float xn  = dv * rsqrtf(var + LNEPS) * gamma[t] + beta[t];
    float xe  = xn > 0.f ? xn : expm1f(xn);
    float s0 = blockReduceSum256(xe * Wl2[t * 2 + 0]);
    float s1 = blockReduceSum256(xe * Wl2[t * 2 + 1]);
    float s2 = blockReduceSum256(xe * Wr2[t * 2 + 0]);
    float s3 = blockReduceSum256(xe * Wr2[t * 2 + 1]);
    if (t == 0) {
        xl2[nid * 2 + 0] = s0; xl2[nid * 2 + 1] = s1;
        xr2[nid * 2 + 0] = s2; xr2[nid * 2 + 1] = s3;
    }
}

// ---------------------------------------------------------------------------
// conv2 aggregation: one wave per node; recompute logits (cheap), two passes,
// writes final output (bias folded in). No atomics, no scratch.
// ---------------------------------------------------------------------------
__global__ __launch_bounds__(256) void agg2_kernel(
    const int* __restrict__ rowptr, const int* __restrict__ esrc,
    const float* __restrict__ xl2, const float* __restrict__ xr2,
    const float* __restrict__ att2, const float* __restrict__ b2,
    float* __restrict__ out) {

    const int nid  = blockIdx.x * 8 + (threadIdx.x >> 5);
    const int lane = threadIdx.x & 31;
    if (nid >= NN) return;
    const int start = rowptr[nid], end = rowptr[nid + 1];

    const float r0 = xr2[2 * nid + 0], r1 = xr2[2 * nid + 1];
    const float w0 = att2[0], w1 = att2[1];

    float mx = -3.0e38f;
    for (int i = start + lane; i < end; i += 32) {
        const int s = esrc[i];
        float a0 = xl2[2 * s + 0] + r0; a0 = a0 > 0.f ? a0 : NEG * a0;
        float a1 = xl2[2 * s + 1] + r1; a1 = a1 > 0.f ? a1 : NEG * a1;
        mx = fmaxf(mx, a0 * w0 + a1 * w1);
    }
    #pragma unroll
    for (int off = 16; off > 0; off >>= 1) mx = fmaxf(mx, __shfl_xor(mx, off, 32));

    float z = 0.f, s0 = 0.f, s1 = 0.f;
    for (int i = start + lane; i < end; i += 32) {
        const int s = esrc[i];
        const float l0 = xl2[2 * s + 0], l1 = xl2[2 * s + 1];
        float a0 = l0 + r0; a0 = a0 > 0.f ? a0 : NEG * a0;
        float a1 = l1 + r1; a1 = a1 > 0.f ? a1 : NEG * a1;
        const float p = __expf(a0 * w0 + a1 * w1 - mx);
        z += p; s0 += p * l0; s1 += p * l1;
    }
    #pragma unroll
    for (int off = 16; off > 0; off >>= 1) {
        z  += __shfl_xor(z,  off, 32);
        s0 += __shfl_xor(s0, off, 32);
        s1 += __shfl_xor(s1, off, 32);
    }
    if (lane == 0) {
        out[2 * nid + 0] = b2[0] + s0 / z;
        out[2 * nid + 1] = b2[1] + s1 / z;
    }
}

// ---------------------------------------------------------------------------
// launch
// ---------------------------------------------------------------------------
extern "C" void kernel_launch(void* const* d_in, const int* in_sizes, int n_in,
                              void* d_out, int out_size, void* d_ws, size_t ws_size,
                              hipStream_t stream) {
    const float* x     = (const float*)d_in[0];
    const int*   ei    = (const int*)  d_in[1];
    const float* Wl1   = (const float*)d_in[2];
    const float* Wr1   = (const float*)d_in[3];
    const float* att1  = (const float*)d_in[4];
    const float* b1    = (const float*)d_in[5];
    const float* Ws    = (const float*)d_in[6];
    const float* bs    = (const float*)d_in[7];
    const float* gamma = (const float*)d_in[8];
    const float* beta  = (const float*)d_in[9];
    const float* Wl2   = (const float*)d_in[10];
    const float* Wr2   = (const float*)d_in[11];
    const float* att2  = (const float*)d_in[12];
    const float* b2    = (const float*)d_in[13];
    float*       out   = (float*)d_out;

    const int E  = in_sizes[1] / 2;   // 320000
    const int ET = E + NN;            // + self loops

    // workspace layout (floats, 16-float = 64B aligned chunks)
    float* base = (float*)d_ws;
    size_t o = 0;
    float*  xl1      = base + o; o += (size_t)NN * HID;
    float*  xr1      = base + o; o += (size_t)NN * HID;
    float*  xs       = base + o; o += (size_t)NN * HID;
    float*  acc      = base + o; o += (size_t)NN * HID;
    __bf16* WT       = (__bf16*)(base + o); o += (size_t)3 * HID * DD / 2;
    float*  lscratch = base + o; o += ((size_t)ET * HH + 15) / 16 * 16;
    int*    rowptr   = (int*)(base + o); o += ((size_t)NN + 16) / 16 * 16;
    int*    woff     = (int*)(base + o); o += (size_t)NN;
    int*    deg      = (int*)(base + o); o += (size_t)NN;
    int*    esrc     = (int*)(base + o); o += ((size_t)ET + 15) / 16 * 16;
    float*  xl2      = base + o; o += (size_t)NN * 2;
    float*  xr2      = base + o; o += (size_t)NN * 2;
    (void)ws_size; (void)n_in; (void)out_size;

    const int T = 256;
    const int gE = (ET + T - 1) / T;

    prep_kernel<<<(3 * HID * DD + T - 1) / T, T, 0, stream>>>(Wl1, Wr1, Ws, WT, deg);
    deg_kernel <<<gE, T, 0, stream>>>(ei, E, ET, deg);
    scan_kernel<<<1, 1024, 0, stream>>>(deg, rowptr, woff);
    fill_kernel<<<gE, T, 0, stream>>>(ei, E, ET, woff, esrc);

    gemm3_kernel<<<dim3(NN / 16, (3 * HID) / 128), T, 0, stream>>>(
        x, WT, bs, xl1, xr1, xs);

    agg1_kernel<<<(NN * HH + 7) / 8, T, 0, stream>>>(
        rowptr, esrc, xl1, xr1, att1, lscratch, acc);

    nodepost_kernel<<<NN, T, 0, stream>>>(acc, xs, b1, gamma, beta, Wl2, Wr2, xl2, xr2);

    agg2_kernel<<<(NN + 7) / 8, T, 0, stream>>>(
        rowptr, esrc, xl2, xr2, att2, b2, out);
}
